// ConvMod_38835094290729
// MI455X (gfx1250) — matrix operand order repaired and checked by reading, not compiled, not run on hardware
//
#include <hip/hip_runtime.h>
#include <math.h>

typedef unsigned short u16;
typedef unsigned int   u32;
typedef unsigned long long u64;
typedef __attribute__((ext_vector_type(16))) __bf16 v16bf;
typedef __attribute__((ext_vector_type(8)))  float  v8f;
typedef __attribute__((ext_vector_type(4)))  u32    u32x4;

union Frag16 { v16bf v; u32x4 q[2]; };

__device__ __forceinline__ u16 f2bf(float x) {
    u32 u = __float_as_uint(x);
    u32 r = u + 0x7FFFu + ((u >> 16) & 1u);
    return (u16)(r >> 16);
}

// A fragment: 16x32 bf16 row-major, contiguous K (matches B):
// lane<16: row=lane, K=0..15; lane>=16: K=16..31
__device__ __forceinline__ v16bf loadA(const u16* __restrict__ p, int hf) {
    Frag16 f;
    f.q[0] = *(const u32x4*)(p + hf * 16);
    f.q[1] = *(const u32x4*)(p + hf * 16 + 8);
    return f.v;
}
// B fragment: 32x16 bf16, X stored position-major [l][c]:
// lane<16: col=lane, K=0..15 contiguous; lane>=16: col=lane-16, K=16..31
__device__ __forceinline__ v16bf loadB(const u16* __restrict__ p) {
    Frag16 f;
    f.q[0] = ((const u32x4*)p)[0];
    f.q[1] = ((const u32x4*)p)[1];
    return f.v;
}

__device__ __forceinline__ v8f wmma_bf16(v16bf a, v16bf b, v8f c) {
    return __builtin_amdgcn_wmma_f32_16x16x32_bf16(false, a, false, b, (short)0, c, false, false);
}

__device__ __forceinline__ float gelu_exact(float x) {
    return 0.5f * x * (1.0f + erff(x * 0.70710678118654752440f));
}

__global__ void cvt_bf16(const float* __restrict__ s, u16* __restrict__ d, int n) {
    int i = blockIdx.x * 256 + threadIdx.x;
    if (i < n) d[i] = f2bf(s[i]);
}

// x[b][c][l] f32  ->  xbf[b][l][c] bf16 (position-major for B operand)
__global__ void xpose_bf16(const float* __restrict__ x, u16* __restrict__ xbf, int Ld) {
    int tid = blockIdx.x * 256 + threadIdx.x;       // tid = (b*Ld + l)*256 + c
    int c = tid & 255;
    int rest = tid >> 8;
    int l = rest % Ld;
    int b = rest / Ld;
    xbf[tid] = f2bf(x[((size_t)(b * 256 + c)) * Ld + l]);
}

__global__ void zero_f32(float* __restrict__ p, int n) {
    int i = blockIdx.x * 256 + threadIdx.x;
    if (i < n) p[i] = 0.0f;
}

enum EpiMode { EPI_GELU_BF16 = 0, EPI_F32_LC = 1, EPI_F32_CL = 2, EPI_AV_BF16 = 3, EPI_ADD_CL = 4 };

// out[m, b, l] = sum_k W[m,k] * X[b][l][k]   (M = Kdim = 256)
// block = 256 thr = 8 waves; wave tile 32(M) x 32(N); block tile 64(M) x 128(N).
// W block-tile is async-staged into LDS (CDNA5 ASYNCcnt path); B double-buffered from global.
template<int EPI>
__launch_bounds__(256)
__global__ void gemm256(const u16* __restrict__ W, const u16* __restrict__ X,
                        const float* __restrict__ bias,
                        float* __restrict__ outF, u16* __restrict__ outBf,
                        const float* __restrict__ vF, int Ld)
{
    __shared__ u16 s_w[64 * 256];                 // 32 KB weight tile

    const int tid  = threadIdx.x;
    const int lane = tid & 31;
    const int wave = tid >> 5;
    const int hf = lane >> 4;
    const int ln = lane & 15;
    const int wm = wave & 1;
    const int wn = wave >> 1;
    const int n0 = blockIdx.x * 128 + wn * 32;
    const int b  = blockIdx.z;

    // --- async copy of the contiguous 64x256 bf16 W tile into LDS ---
    {
        const u16* src = W + (size_t)blockIdx.y * (64 * 256);
        #pragma unroll
        for (int j = 0; j < 8; ++j) {
            int e = (tid + j * 256) * 8;                       // 16B chunk per lane
            u32 lo = (u32)(uintptr_t)(s_w + e);
            u64 ga = (u64)(uintptr_t)(src + e);
            asm volatile("global_load_async_to_lds_b128 %0, %1, off"
                         :: "v"(lo), "v"(ga) : "memory");
        }
        asm volatile("s_wait_asynccnt 0x0" ::: "memory");
        __syncthreads();
    }

    const u16* A0 = s_w + (size_t)(wm * 32 + ln) * 256;       // mt=0 rows
    const u16* A1 = A0 + 16 * 256;                            // mt=1 rows
    const u16* B0 = X + (size_t)(b * Ld + n0 + ln) * 256 + hf * 16;
    const u16* B1 = B0 + (size_t)16 * 256;

    v8f acc00 = {}, acc01 = {}, acc10 = {}, acc11 = {};
    Frag16 bc0 = {}, bc1 = {}, bn0 = {}, bn1 = {};
    bc0.v = loadB(B0);
    bc1.v = loadB(B1);
    #pragma unroll
    for (int kk = 0; kk < 256; kk += 32) {
        if (kk + 32 < 256) {                                  // prefetch next-K B frags
            bn0.v = loadB(B0 + kk + 32);
            bn1.v = loadB(B1 + kk + 32);
        }
        v16bf a0 = loadA(A0 + kk, hf);                        // ds_load_b128 from LDS
        v16bf a1 = loadA(A1 + kk, hf);
        acc00 = wmma_bf16(a0, bc0.v, acc00);
        acc01 = wmma_bf16(a0, bc1.v, acc01);
        acc10 = wmma_bf16(a1, bc0.v, acc10);
        acc11 = wmma_bf16(a1, bc1.v, acc11);
        bc0 = bn0; bc1 = bn1;
    }

    const int m0g = blockIdx.y * 64 + wm * 32;
    #pragma unroll
    for (int mt = 0; mt < 2; ++mt) {
        const int mb = m0g + mt * 16 + hf * 8;                // 8 consecutive out channels
        #pragma unroll
        for (int nt = 0; nt < 2; ++nt) {
            v8f acc = (mt == 0) ? (nt == 0 ? acc00 : acc01)
                                : (nt == 0 ? acc10 : acc11);
            const int gl = n0 + nt * 16 + ln;
            if constexpr (EPI == EPI_GELU_BF16 || EPI == EPI_AV_BF16) {
                const size_t ob = (size_t)(b * Ld + gl) * 256 + mb;
                float vv[8];
                #pragma unroll
                for (int r = 0; r < 8; ++r) {
                    float v = acc[r] + bias[mb + r];
                    if constexpr (EPI == EPI_GELU_BF16) v = gelu_exact(v);
                    else                                v = v * vF[ob + r];
                    vv[r] = v;
                }
                u32x4 st;
                st.x = (u32)f2bf(vv[0]) | ((u32)f2bf(vv[1]) << 16);
                st.y = (u32)f2bf(vv[2]) | ((u32)f2bf(vv[3]) << 16);
                st.z = (u32)f2bf(vv[4]) | ((u32)f2bf(vv[5]) << 16);
                st.w = (u32)f2bf(vv[6]) | ((u32)f2bf(vv[7]) << 16);
                *(u32x4*)(outBf + ob) = st;
            } else if constexpr (EPI == EPI_F32_LC) {
                const size_t ob = (size_t)(b * Ld + gl) * 256 + mb;
                #pragma unroll
                for (int r = 0; r < 8; ++r) outF[ob + r] = acc[r] + bias[mb + r];
            } else if constexpr (EPI == EPI_F32_CL) {
                #pragma unroll
                for (int r = 0; r < 8; ++r)
                    outF[(size_t)(b * 256 + mb + r) * Ld + gl] = acc[r] + bias[mb + r];
            } else { // EPI_ADD_CL
                #pragma unroll
                for (int r = 0; r < 8; ++r) {
                    size_t o = (size_t)(b * 256 + mb + r) * Ld + gl;
                    outF[o] += acc[r] + bias[mb + r];
                }
            }
        }
    }
}

// Fused deformable depthwise: per block = (16 positions) x (16 channels) x batch.
// Phase 1: WMMA GEMMs for off & msk tiles (rows c*K+k of off_w/msk_w) -> LDS (f32).
// Phase 2: softmax over K + bilinear gather from xin (f32, L2-resident) -> dcn bf16.
#define KMAXC 17
__launch_bounds__(256)
__global__ void dcn_kernel(const u16* __restrict__ offW, const u16* __restrict__ mskW,
                           const float* __restrict__ offB, const float* __restrict__ mskB,
                           const u16* __restrict__ a1, const float* __restrict__ xin,
                           u16* __restrict__ dcn, int K, int Ld)
{
    __shared__ float s_off[16 * KMAXC * 16];
    __shared__ float s_msk[16 * KMAXC * 16];

    const int lane = threadIdx.x & 31;
    const int wave = threadIdx.x >> 5;
    const int hf = lane >> 4;
    const int ln = lane & 15;
    const int l0 = blockIdx.x * 16;
    const int c0 = blockIdx.y * 16;
    const int b  = blockIdx.z;
    const int row0 = c0 * K;                 // first off_w/msk_w row for this channel tile

    // preload all 8 B fragments (a1 tile) once; invariant over the t loop
    Frag16 bfr[8];
    {
        const u16* Bp = a1 + (size_t)(b * Ld + l0 + ln) * 256 + hf * 16;
        #pragma unroll
        for (int j = 0; j < 8; ++j) bfr[j].v = loadB(Bp + j * 32);
    }

    // 2K m-tiles of 16 rows each (off then msk), spread over the 8 waves
    for (int t = wave; t < 2 * K; t += 8) {
        const bool isM = (t >= K);
        const int mt = isM ? t - K : t;
        const u16* Wp = (isM ? mskW : offW) + (size_t)(row0 + mt * 16 + ln) * 256;
        const float* bp = isM ? mskB : offB;

        Frag16 afr[8];                          // batch-issue all A loads, then drain WMMAs
        #pragma unroll
        for (int j = 0; j < 8; ++j) {
            afr[j].q[0] = *(const u32x4*)(Wp + j * 32 + hf * 16);
            afr[j].q[1] = *(const u32x4*)(Wp + j * 32 + hf * 16 + 8);
        }
        v8f acc = {};
        #pragma unroll
        for (int j = 0; j < 8; ++j) acc = wmma_bf16(afr[j].v, bfr[j].v, acc);

        float* s = isM ? s_msk : s_off;
        #pragma unroll
        for (int r = 0; r < 8; ++r) {
            int lr = mt * 16 + r + hf * 8;      // local row = c_local*K + k
            s[lr * 16 + ln] = acc[r] + bp[row0 + lr];
        }
    }
    __syncthreads();

    // one thread per (position, channel); consecutive threads -> consecutive channels
    const int cl = threadIdx.x & 15;
    const int ll = threadIdx.x >> 4;
    const int l = l0 + ll;
    const int c = c0 + cl;
    const int base = cl * K;
    const int center = (K - 1) >> 1;

    float mx = -1e30f;
    for (int k = 0; k < K; ++k) mx = fmaxf(mx, s_msk[(base + k) * 16 + ll]);
    float sum = 0.0f;
    for (int k = 0; k < K; ++k) sum += __expf(s_msk[(base + k) * 16 + ll] - mx);
    const float rs = 1.0f / sum;

    const float* xr = xin + (size_t)(b * 256 + c) * Ld;
    float accv = 0.0f;
    for (int k = 0; k < K; ++k) {
        float pk  = __expf(s_msk[(base + k) * 16 + ll] - mx) * rs;
        float off = s_off[(base + k) * 16 + ll];
        float pos = (float)(l + k - center) + off;
        float i0f = floorf(pos);
        float w1 = pos - i0f;
        float w0 = 1.0f - w1;
        int i0 = (int)i0f;
        int i1 = i0 + 1;
        float g0 = (i0 >= 0 && i0 < Ld) ? xr[i0] : 0.0f;
        float g1 = (i1 >= 0 && i1 < Ld) ? xr[i1] : 0.0f;
        accv += pk * (w0 * g0 + w1 * g1);
    }
    dcn[(size_t)(b * Ld + l) * 256 + c] = f2bf(accv);
}

extern "C" void kernel_launch(void* const* d_in, const int* in_sizes, int n_in,
                              void* d_out, int out_size, void* d_ws, size_t ws_size,
                              hipStream_t stream)
{
    (void)in_sizes; (void)n_in; (void)ws_size;
    const int Bd = 4, Cd = 256, Ld = 2048, Pd = 6, KM = 17;

    const float* x     = (const float*)d_in[0];
    const float* a_w   = (const float*)d_in[1];
    const float* a_b   = (const float*)d_in[2];
    const float* v_w   = (const float*)d_in[3];
    const float* v_b   = (const float*)d_in[4];
    const float* o_w   = (const float*)d_in[5];
    const float* o_b   = (const float*)d_in[6];
    const float* in_w  = (const float*)d_in[7];
    const float* in_b  = (const float*)d_in[8];
    const float* ow_w  = (const float*)d_in[9];
    const float* ow_b  = (const float*)d_in[10];
    const float* off_w = (const float*)d_in[11];
    const float* off_b = (const float*)d_in[12];
    const float* msk_w = (const float*)d_in[13];
    const float* msk_b = (const float*)d_in[14];
    float* out = (float*)d_out;

    char* ws = (char*)d_ws;
    size_t cur = 0;
    auto alloc = [&](size_t bytes) -> void* {
        void* p = ws + cur;
        cur += (bytes + 255) & ~(size_t)255;
        return p;
    };
    const size_t nAct = (size_t)Bd * Ld * Cd;                 // 2,097,152
    u16*   x_bf   = (u16*)alloc(nAct * 2);
    u16*   a1_bf  = (u16*)alloc(nAct * 2);
    u16*   dcn_bf = (u16*)alloc(nAct * 2);
    u16*   av_bf  = (u16*)alloc(nAct * 2);
    float* v_f    = (float*)alloc(nAct * 4);
    float* xin_f  = (float*)alloc(nAct * 4);
    const size_t nWsq = (size_t)Pd * Cd * Cd;                 // 393,216
    u16* aWb  = (u16*)alloc(nWsq * 2);
    u16* vWb  = (u16*)alloc(nWsq * 2);
    u16* oWb  = (u16*)alloc(nWsq * 2);
    u16* inWb = (u16*)alloc(nWsq * 2);
    u16* owWb = (u16*)alloc(nWsq * 2);
    const size_t nWk = (size_t)Pd * Cd * KM * Cd;             // 6,684,672
    u16* offWb = (u16*)alloc(nWk * 2);
    u16* mskWb = (u16*)alloc(nWk * 2);

    const int T = 256;
    cvt_bf16<<<dim3((unsigned)((nWsq + T - 1) / T)), T, 0, stream>>>(a_w,  aWb,  (int)nWsq);
    cvt_bf16<<<dim3((unsigned)((nWsq + T - 1) / T)), T, 0, stream>>>(v_w,  vWb,  (int)nWsq);
    cvt_bf16<<<dim3((unsigned)((nWsq + T - 1) / T)), T, 0, stream>>>(o_w,  oWb,  (int)nWsq);
    cvt_bf16<<<dim3((unsigned)((nWsq + T - 1) / T)), T, 0, stream>>>(in_w, inWb, (int)nWsq);
    cvt_bf16<<<dim3((unsigned)((nWsq + T - 1) / T)), T, 0, stream>>>(ow_w, owWb, (int)nWsq);
    cvt_bf16<<<dim3((unsigned)((nWk + T - 1) / T)),  T, 0, stream>>>(off_w, offWb, (int)nWk);
    cvt_bf16<<<dim3((unsigned)((nWk + T - 1) / T)),  T, 0, stream>>>(msk_w, mskWb, (int)nWk);
    xpose_bf16<<<dim3((unsigned)((nAct + T - 1) / T)), T, 0, stream>>>(x, x_bf, Ld);
    zero_f32<<<dim3((unsigned)((out_size + T - 1) / T)), T, 0, stream>>>(out, out_size);

    dim3 gg(Ld / 128, Cd / 64, Bd);
    dim3 gd(Ld / 16, Cd / 16, Bd);
    for (int i = 0; i < Pd; ++i) {
        const int K = 7 + 2 * i;
        const size_t wq = (size_t)i * Cd * Cd;
        const size_t wk = (size_t)i * Cd * KM * Cd;
        // a1 = gelu(a_w @ x) -> bf16 [b][l][c]
        gemm256<EPI_GELU_BF16><<<gg, T, 0, stream>>>(aWb + wq, x_bf, a_b + i * Cd, nullptr, a1_bf, nullptr, Ld);
        // v = v_w @ x -> f32 [b][l][c]
        gemm256<EPI_F32_LC><<<gg, T, 0, stream>>>(vWb + wq, x_bf, v_b + i * Cd, v_f, nullptr, nullptr, Ld);
        // xin = in_w @ a1 -> f32 [b][c][l] (gather source)
        gemm256<EPI_F32_CL><<<gg, T, 0, stream>>>(inWb + wq, a1_bf, in_b + i * Cd, xin_f, nullptr, nullptr, Ld);
        // fused off/msk GEMM + softmax + bilinear gather -> dcn bf16 [b][l][c]
        dcn_kernel<<<gd, T, 0, stream>>>(offWb + wk, mskWb + wk,
                                         off_b + (size_t)i * Cd * KM, msk_b + (size_t)i * Cd * KM,
                                         a1_bf, xin_f, dcn_bf, K, Ld);
        // av = (ow_w @ dcn + ow_b) * v -> bf16 [b][l][c]
        gemm256<EPI_AV_BF16><<<gg, T, 0, stream>>>(owWb + wq, dcn_bf, ow_b + i * Cd, nullptr, av_bf, v_f, Ld);
        // out += o_w @ av + o_b  (f32, [b][c][l])
        gemm256<EPI_ADD_CL><<<gg, T, 0, stream>>>(oWb + wq, av_bf, o_b + i * Cd, out, nullptr, nullptr, Ld);
    }
}
